// Attention_update_85658827751963
// MI455X (gfx1250) — matrix operand order repaired and checked by reading, not compiled, not run on hardware
//
#include <hip/hip_runtime.h>
#include <math.h>

typedef __attribute__((ext_vector_type(16))) __bf16 v16bf;
typedef __attribute__((ext_vector_type(8)))  float  v8f;
typedef __attribute__((ext_vector_type(4)))  unsigned int v4u;
typedef __attribute__((ext_vector_type(8)))  int v8i;
typedef __attribute__((ext_vector_type(4)))  int v4i;

#define B_  64
#define S_  2048
#define DK_ 1024
#define D_  1024

__device__ __forceinline__ unsigned short f2bf(float f) {
  unsigned int u = __float_as_uint(f);
  u += 0x7FFFu + ((u >> 16) & 1u);        // round-to-nearest-even
  return (unsigned short)(u >> 16);
}

__device__ __forceinline__ float fast_tanh(float x) {
  return 1.0f - 2.0f / (1.0f + __expf(2.0f * x));
}

// ---------------- W1e f32 -> bf16 (4 MB -> 2 MB, L2-resident afterwards) ----
__global__ __launch_bounds__(256) void k_convert_w1e(const float* __restrict__ W1,
                                                     unsigned short* __restrict__ w1e) {
  int i4 = blockIdx.x * 256 + threadIdx.x;   // 262144 float4's
  int d  = i4 >> 8;
  int k  = (i4 & 255) * 4;
  float4 f = *(const float4*)(W1 + (size_t)d * 2048 + k);
  ushort4 h;
  h.x = f2bf(f.x); h.y = f2bf(f.y); h.z = f2bf(f.z); h.w = f2bf(f.w);
  *(ushort4*)(w1e + (size_t)d * 1024 + k) = h;
}

// ---------------- hproj[b,d] = b1[d] + hidden[b,:] @ W1h[d,:]  (f32, m-split)
__global__ __launch_bounds__(256) void k_hproj_partial(const float* __restrict__ hidden,
                                                       const float* __restrict__ W1,
                                                       float* __restrict__ hpp) {
  int mc = blockIdx.x >> 2;                       // 8 m-chunks of 128
  int d  = (blockIdx.x & 3) * 256 + threadIdx.x;  // 1024 d's
  float acc[64];
#pragma unroll
  for (int b = 0; b < 64; ++b) acc[b] = 0.f;
  int m0 = mc * 128;
  for (int m = m0; m < m0 + 128; ++m) {
    float wv = W1[(size_t)d * 2048 + 1024 + m];   // W1h[d][m]
#pragma unroll
    for (int b = 0; b < 64; ++b) acc[b] = fmaf(hidden[b * 1024 + m], wv, acc[b]);
  }
  for (int b = 0; b < 64; ++b) hpp[((size_t)mc * 64 + b) * 1024 + d] = acc[b];
}

__global__ __launch_bounds__(256) void k_hproj_reduce(const float* __restrict__ hpp,
                                                      const float* __restrict__ b1,
                                                      float* __restrict__ hproj) {
  int idx = blockIdx.x * 256 + threadIdx.x;       // 65536 = 64*1024
  int d   = idx & 1023;
  float s = b1[d];
#pragma unroll
  for (int mc = 0; mc < 8; ++mc) s += hpp[(size_t)mc * 65536 + idx];
  hproj[idx] = s;
}

// ---- TDM: DMA one 128(d) x 128(k) bf16 tile of W1e into LDS (padded rows) --
// D# per CDNA5 ISA ch.8: group0 = {flags, lds_addr, global_addr, type=2},
// group1 = {data_size=2B, pad: +4 DWORDs after every 64 DWORDs (-> 272B rows),
//           tensor 1024x1024, tile 128x128, dim0_stride 1024}
__device__ __forceinline__ void tdm_load_w_tile(const unsigned short* w1e,
                                                unsigned lds_base, int c) {
  int nt = c >> 3;
  int kc = (c & 7) * 128;
  unsigned long long ga =
      (unsigned long long)(size_t)(w1e + ((size_t)nt * 128) * 1024 + kc);
  v4u g0;
  g0[0] = 1u;                                        // count=1, no gather
  g0[1] = lds_base;                                  // lds_addr (bytes)
  g0[2] = (unsigned)(ga & 0xFFFFFFFFu);              // global_addr[31:0]
  g0[3] = (unsigned)((ga >> 32) & 0x01FFFFFFu)       // global_addr[56:32]
        | (2u << 30);                                // type = 2 ("image")
  v8i g1;
  g1[0] = (int)((1u << 16)                           // data_size = 2 bytes
              | (1u << 20)                           // pad_enable
              | (5u << 22)                           // pad_interval: 64 DWORDs
              | (3u << 25));                         // pad_amount: 4 DWORDs
  g1[1] = (int)((1024u & 0xFFFFu) << 16);            // tensor_dim0[15:0]
  g1[2] = (int)(((1024u >> 16) & 0xFFFFu)            // tensor_dim0[31:16]
              | ((1024u & 0xFFFFu) << 16));          // tensor_dim1[15:0]
  g1[3] = (int)(((1024u >> 16) & 0xFFFFu)            // tensor_dim1[31:16]
              | (128u << 16));                       // tile_dim0 = 128
  g1[4] = (int)(128u);                               // tile_dim1 = 128, tile_dim2 = 0
  g1[5] = (int)1024;                                 // tensor_dim0_stride[31:0]
  g1[6] = 0;                                         // stride hi / dim1_stride lo
  g1[7] = 0;
  v4i z4 = {0, 0, 0, 0};
  v8i z8 = {0, 0, 0, 0, 0, 0, 0, 0};
  __builtin_amdgcn_tensor_load_to_lds(g0, g1, z4, z4, z8, 0);
}

// ---------------- fused  e[b,s] = sum_d tanh(enc@W1e^T + hproj)*w2  ---------
// One workgroup = 64 s-rows (one b); 8 waves; N swept in 8 tiles of 128.
// Each wave owns a 2x2 block of 16x16 WMMA tiles -> 2 ds_load_b128 per WMMA.
// W1e tiles arrive via double-buffered TENSOR_LOAD_TO_LDS issued by wave 0.
__global__ __launch_bounds__(256) void k_attn_e(const float* __restrict__ enc,
                                                const unsigned short* __restrict__ w1e,
                                                const float* __restrict__ hproj,
                                                const float* __restrict__ W2,
                                                float* __restrict__ e_out) {
  __shared__ unsigned short As[64 * 1032];       // 64 x 1024 bf16, +8 pad/row
  __shared__ unsigned short Ws[2][128 * 136];    // double-buffered W tiles
  __shared__ float e_red[4][64];

  const int wg = blockIdx.x;
  const int b  = wg >> 5;
  const int s0 = (wg & 31) << 6;
  const float* encA = enc + ((size_t)b * S_ + s0) * DK_;

  const int tid = threadIdx.x;
  const int wave = tid >> 5, lane = tid & 31;
  const int half = lane >> 4, l15 = lane & 15;

  // kick off DMA of first W tile while we stage the A panel
  if (wave == 0) tdm_load_w_tile(w1e, (unsigned)(size_t)&Ws[0][0], 0);

  // Stage full A panel: 64 rows x 1024 k, f32 -> bf16 (coalesced 16B loads)
#pragma unroll 2
  for (int j = 0; j < 64; ++j) {
    float4 f = *(const float4*)(encA + (size_t)j * 1024 + tid * 4);
    ushort4 h;
    h.x = f2bf(f.x); h.y = f2bf(f.y); h.z = f2bf(f.z); h.w = f2bf(f.w);
    *(ushort4*)(&As[j * 1032 + tid * 4]) = h;
  }

  const int mb   = (wave & 1) * 32;        // two 16-row m-blocks: mb, mb+16
  const int nbq  = (wave >> 1) * 32;       // two 16-col n-blocks: nbq, nbq+16
  const int arow0 = (mb + l15) * 1032;
  const int arow1 = (mb + 16 + l15) * 1032;
  const int brow0 = (nbq + l15) * 136;
  const int brow1 = (nbq + 16 + l15) * 136;

  float ef0[8], ef1[8];
#pragma unroll
  for (int v = 0; v < 8; ++v) { ef0[v] = 0.f; ef1[v] = 0.f; }

  for (int nt = 0; nt < 8; ++nt) {
    v8f acc00 = {}, acc01 = {}, acc10 = {}, acc11 = {};

    for (int kci = 0; kci < 8; ++kci) {
      const int c = nt * 8 + kci;
      const unsigned short* WsB = &Ws[c & 1][0];

      if (wave == 0) {                       // prefetch next tile, wait current
        if (c + 1 < 64) {
          tdm_load_w_tile(w1e, (unsigned)(size_t)&Ws[(c + 1) & 1][0], c + 1);
          __builtin_amdgcn_s_wait_tensorcnt(1);
        } else {
          __builtin_amdgcn_s_wait_tensorcnt(0);
        }
      }
      __syncthreads();                       // tile c visible to all waves

#pragma unroll
      for (int kk = 0; kk < 128; kk += 32) {
        union { v16bf v; uint4 q[2]; } a0, a1, b0, b1;
        int ka = kci * 128 + kk + half * 8;  // A frag K pattern {+0..7,+16..23}
        a0.q[0] = *(const uint4*)(&As[arow0 + ka]);
        a0.q[1] = *(const uint4*)(&As[arow0 + ka + 16]);
        a1.q[0] = *(const uint4*)(&As[arow1 + ka]);
        a1.q[1] = *(const uint4*)(&As[arow1 + ka + 16]);
        int kb = kk + half * 8;              // B frag K pattern {+0..7,+16..23}
        b0.q[0] = *(const uint4*)(&WsB[brow0 + kb]);
        b0.q[1] = *(const uint4*)(&WsB[brow0 + kb + 16]);
        b1.q[0] = *(const uint4*)(&WsB[brow1 + kb]);
        b1.q[1] = *(const uint4*)(&WsB[brow1 + kb + 16]);
        acc00 = __builtin_amdgcn_wmma_f32_16x16x32_bf16(false, a0.v, false, b0.v,
                                                        (short)0, acc00, false, false);
        acc01 = __builtin_amdgcn_wmma_f32_16x16x32_bf16(false, a0.v, false, b1.v,
                                                        (short)0, acc01, false, false);
        acc10 = __builtin_amdgcn_wmma_f32_16x16x32_bf16(false, a1.v, false, b0.v,
                                                        (short)0, acc10, false, false);
        acc11 = __builtin_amdgcn_wmma_f32_16x16x32_bf16(false, a1.v, false, b1.v,
                                                        (short)0, acc11, false, false);
      }
      __syncthreads();                       // all reads of tile c done (WAR)
    }

    // fused epilogue: pre = acc + hproj ; ef += tanh(pre)*w2
    int d0 = nt * 128 + nbq + l15;
    int d1 = d0 + 16;
    float hp0 = hproj[b * 1024 + d0], w20 = W2[d0];
    float hp1 = hproj[b * 1024 + d1], w21 = W2[d1];
#pragma unroll
    for (int v = 0; v < 8; ++v) {
      ef0[v] += fast_tanh(acc00[v] + hp0) * w20 + fast_tanh(acc01[v] + hp1) * w21;
      ef1[v] += fast_tanh(acc10[v] + hp0) * w20 + fast_tanh(acc11[v] + hp1) * w21;
    }
  }

  // butterfly-reduce the 16 lanes sharing a row (halves stay disjoint)
#pragma unroll
  for (int v = 0; v < 8; ++v) {
    float x = ef0[v], y = ef1[v];
    x += __shfl_xor(x, 1, 32);  y += __shfl_xor(y, 1, 32);
    x += __shfl_xor(x, 2, 32);  y += __shfl_xor(y, 2, 32);
    x += __shfl_xor(x, 4, 32);  y += __shfl_xor(y, 4, 32);
    x += __shfl_xor(x, 8, 32);  y += __shfl_xor(y, 8, 32);
    ef0[v] = x; ef1[v] = y;
  }
  if (l15 == 0) {
    const int slot = wave >> 1;              // one slot per n-quarter
#pragma unroll
    for (int v = 0; v < 8; ++v) {
      e_red[slot][mb + (half << 3) + v]      = ef0[v];
      e_red[slot][mb + 16 + (half << 3) + v] = ef1[v];
    }
  }
  __syncthreads();
  if (tid < 64)
    e_out[(size_t)b * S_ + s0 + tid] =
        (e_red[0][tid] + e_red[1][tid]) + (e_red[2][tid] + e_red[3][tid]);
}

// ---------------- softmax over S per batch row ------------------------------
__global__ __launch_bounds__(256) void k_softmax(const float* __restrict__ e,
                                                 float* __restrict__ alpha) {
  __shared__ float red[256];
  int b = blockIdx.x, tid = threadIdx.x;
  const float* er = e + (size_t)b * S_;
  float v[8], mx = -INFINITY;
#pragma unroll
  for (int j = 0; j < 8; ++j) { v[j] = er[tid + j * 256]; mx = fmaxf(mx, v[j]); }
  red[tid] = mx; __syncthreads();
  for (int off = 128; off > 0; off >>= 1) {
    if (tid < off) red[tid] = fmaxf(red[tid], red[tid + off]);
    __syncthreads();
  }
  mx = red[0]; __syncthreads();
  float s = 0.f;
#pragma unroll
  for (int j = 0; j < 8; ++j) { v[j] = __expf(v[j] - mx); s += v[j]; }
  red[tid] = s; __syncthreads();
  for (int off = 128; off > 0; off >>= 1) {
    if (tid < off) red[tid] += red[tid + off];
    __syncthreads();
  }
  float inv = 1.0f / red[0];
#pragma unroll
  for (int j = 0; j < 8; ++j) alpha[(size_t)b * S_ + tid + j * 256] = v[j] * inv;
}

// ---------------- context[b,k] = sum_s alpha[b,s]*enc[b,s,k]  (s-split) -----
__global__ __launch_bounds__(256) void k_ctx_partial(const float* __restrict__ enc,
                                                     const float* __restrict__ alpha,
                                                     float* __restrict__ cp) {
  int b = blockIdx.x >> 3, sc = blockIdx.x & 7;
  int k = threadIdx.x * 4;
  const float* eb = enc + ((size_t)b * S_ + sc * 256) * DK_ + k;
  const float* ab = alpha + (size_t)b * S_ + sc * 256;
  float a0 = 0.f, a1 = 0.f, a2 = 0.f, a3 = 0.f;
  for (int s = 0; s < 256; ++s) {
    float al = ab[s];
    float4 x = *(const float4*)(eb + (size_t)s * DK_);
    a0 = fmaf(al, x.x, a0); a1 = fmaf(al, x.y, a1);
    a2 = fmaf(al, x.z, a2); a3 = fmaf(al, x.w, a3);
  }
  float4 r; r.x = a0; r.y = a1; r.z = a2; r.w = a3;
  *(float4*)(cp + ((size_t)sc * 64 + b) * DK_ + k) = r;
}

__global__ __launch_bounds__(256) void k_ctx_reduce(const float* __restrict__ cp,
                                                    float* __restrict__ out) {
  int idx = blockIdx.x * 256 + threadIdx.x;   // 65536 = 64*1024
  float s = 0.f;
#pragma unroll
  for (int sc = 0; sc < 8; ++sc) s += cp[(size_t)sc * 65536 + idx];
  out[idx] = s;
}

// ---------------------------------------------------------------------------
extern "C" void kernel_launch(void* const* d_in, const int* in_sizes, int n_in,
                              void* d_out, int out_size, void* d_ws, size_t ws_size,
                              hipStream_t stream) {
  const float* hidden = (const float*)d_in[0];   // (64,1024)
  const float* enc    = (const float*)d_in[1];   // (64,2048,1024)
  const float* W1     = (const float*)d_in[2];   // (1024,2048)
  const float* b1     = (const float*)d_in[3];   // (1024,)
  const float* W2     = (const float*)d_in[4];   // (1,1024)
  float* out = (float*)d_out;                    // (64,1024)

  char* ws = (char*)d_ws;
  size_t o = 0;
  unsigned short* w1e = (unsigned short*)(ws + o); o += (size_t)1024 * 1024 * 2; // 2 MB
  float* hpp   = (float*)(ws + o); o += (size_t)8 * 64 * 1024 * 4;               // 2 MB
  float* hproj = (float*)(ws + o); o += (size_t)64 * 1024 * 4;                   // 256 KB
  float* e     = (float*)(ws + o); o += (size_t)64 * 2048 * 4;                   // 512 KB
  float* alpha = (float*)(ws + o); o += (size_t)64 * 2048 * 4;                   // 512 KB
  float* cp    = (float*)(ws + o); o += (size_t)8 * 64 * 1024 * 4;               // 2 MB

  k_convert_w1e  <<<1024, 256, 0, stream>>>(W1, w1e);
  k_hproj_partial<<<32,   256, 0, stream>>>(hidden, W1, hpp);
  k_hproj_reduce <<<256,  256, 0, stream>>>(hpp, b1, hproj);
  k_attn_e       <<<2048, 256, 0, stream>>>(enc, w1e, hproj, W2, e);
  k_softmax      <<<64,   256, 0, stream>>>(e, alpha);
  k_ctx_partial  <<<512,  256, 0, stream>>>(enc, alpha, cp);
  k_ctx_reduce   <<<256,  256, 0, stream>>>(cp, out);
}
